// Find_31885837205924
// MI455X (gfx1250) — matrix operand order repaired and checked by reading, not compiled, not run on hardware
//
#include <hip/hip_runtime.h>
#include <hip/hip_bf16.h>

// Problem constants
#define BB 16
#define LL 4095
#define DD 768
#define TT 768
#define SS 64
#define KTOP 8
#define HH 384
#define G3 1152   // 3*H

typedef __attribute__((ext_vector_type(16))) __bf16 v16bf;
typedef __attribute__((ext_vector_type(8)))  float  v8f;

__device__ __forceinline__ __bf16 to_bf16(float f) {
  unsigned u = __builtin_bit_cast(unsigned, f);
  unsigned r = u + 0x7fffu + ((u >> 16) & 1u);   // round-to-nearest-even
  unsigned short h = (unsigned short)(r >> 16);
  return __builtin_bit_cast(__bf16, h);
}

__device__ __forceinline__ v8f wmma_bf16(v16bf a, v16bf b, v8f c) {
  // D = A(16x32) * B(32x16) + C, f32 accumulate
  return __builtin_amdgcn_wmma_f32_16x16x32_bf16(false, a, false, b, (short)0, c, false, false);
}

// ---- A fragment: 16x32 bf16 tile of row-major A (lda elems/row), rows 0..15 of A ----
// contiguous k order used here (matches load_fragB): lane l holds row (l&15); K = koff..koff+15, koff = 16*(l>=16)
__device__ __forceinline__ v16bf load_fragA(const __bf16* A, int lda, int kBase, int lane) {
  int m    = lane & 15;
  int koff = (lane >> 4) << 4;
  const __bf16* p = A + (size_t)m * lda + kBase + koff;
  v16bf f;
#pragma unroll
  for (int i = 0; i < 16; ++i) f[i] = p[i];
  return f;
}

// ---- B fragment from W stored (N x K) row-major (B = W^T). Contiguous 16 bf16 per lane. ----
// lane l holds column n = nBase + (l&15); K = kBase + 16*(l>=16) + e
__device__ __forceinline__ v16bf load_fragB(const __bf16* W, int ldw, int nBase, int kBase, int lane) {
  int n  = nBase + (lane & 15);
  int ks = kBase + ((lane >> 4) << 4);
  const __bf16* p = W + (size_t)n * ldw + ks;
  v16bf f;
#pragma unroll
  for (int i = 0; i < 16; ++i) f[i] = p[i];
  return f;
}

// =================== utility kernels ===================

__global__ void cvt_bf16_kernel(const float* __restrict__ src, __bf16* __restrict__ dst, int n) {
  for (int i = blockIdx.x * blockDim.x + threadIdx.x; i < n; i += gridDim.x * blockDim.x)
    dst[i] = to_bf16(src[i]);
}

__global__ void zero_f32_kernel(float* __restrict__ p, int n) {
  for (int i = blockIdx.x * blockDim.x + threadIdx.x; i < n; i += gridDim.x * blockDim.x)
    p[i] = 0.0f;
}

// =================== span averaging: span_rep (B,S,D) as bf16 ===================
__global__ void span_rep_kernel(const float* __restrict__ hidden,
                                const int* __restrict__ span_start,
                                const int* __restrict__ span_len,
                                __bf16* __restrict__ sr_bf) {
  int bs = blockIdx.x;                 // 0..B*S-1
  int b  = bs >> 6;
  int st = span_start[bs];
  int ln = span_len[bs];
  float inv = 1.0f / (float)ln;
  const float* base = hidden + ((size_t)b * LL + st) * DD;
  for (int d = threadIdx.x; d < DD; d += blockDim.x) {
    float acc = 0.0f;
    for (int i = 0; i < ln; ++i) acc += base[(size_t)i * DD + d];
    sr_bf[(size_t)bs * DD + d] = to_bf16(acc * inv);
  }
}

// =================== ctx mean over prefix rows (dominant HBM traffic ~201MB) ===================
__global__ void ctx_partial_kernel(const float* __restrict__ hidden,
                                   const int* __restrict__ attn_len,
                                   float* __restrict__ ctx_acc) {
  int b     = blockIdx.x >> 5;         // 16 batches x 32 row-chunks
  int chunk = blockIdx.x & 31;
  int lim   = attn_len[b] - 1;         // rows [0, lim)
  int r0 = chunk * 128;
  int r1 = r0 + 128; if (r1 > LL) r1 = LL; if (r1 > lim) r1 = lim;
  if (r0 >= r1) return;
  int d = threadIdx.x;                 // 256 threads, 3 columns each
  float a0 = 0.f, a1 = 0.f, a2 = 0.f;
  const float* p = hidden + ((size_t)b * LL + r0) * DD;
  for (int r = r0; r < r1; ++r, p += DD) {
    __builtin_prefetch(p + 4 * DD, 0, 0);     // global_prefetch_b8
    a0 += p[d];
    a1 += p[d + 256];
    a2 += p[d + 512];
  }
  atomicAdd(&ctx_acc[b * DD + d], a0);
  atomicAdd(&ctx_acc[b * DD + d + 256], a1);
  atomicAdd(&ctx_acc[b * DD + d + 512], a2);
}

__global__ void ctx_finalize_kernel(const float* __restrict__ ctx_acc,
                                    const int* __restrict__ attn_len,
                                    __bf16* __restrict__ ctx_bf) {
  int b = blockIdx.x;
  float inv = 1.0f / (float)(attn_len[b] - 1);
  for (int d = threadIdx.x; d < DD; d += blockDim.x)
    ctx_bf[b * DD + d] = to_bf16(ctx_acc[b * DD + d] * inv);
}

// =================== generic bf16 WMMA GEMM: C(MxN) = A(MxK) * W(NxK)^T + bias ===================
// Each wave computes a 16(M) x 64(N) strip. Fragments are double-buffered so the
// loads for k+32 overlap the 4 WMMAs of step k (partial s_wait_loadcnt instead of 0).
__global__ void gemm_bf16_kernel(const __bf16* __restrict__ A, const __bf16* __restrict__ W,
                                 const float* __restrict__ bias, float* __restrict__ C,
                                 int M, int N, int Kd) {
  int lane  = threadIdx.x & 31;
  int wib   = threadIdx.x >> 5;
  int wpb   = blockDim.x >> 5;
  int wave  = blockIdx.x * wpb + wib;
  int numMT = M >> 4;
  int numNG = N >> 6;
  int total = numMT * numNG;
  int nc = lane & 15;
  int m0 = (lane >= 16) ? 8 : 0;
  for (int t = wave; t < total; t += gridDim.x * wpb) {
    int mt = t / numNG, ng = t % numNG;
    int mBase = mt << 4, nBase = ng << 6;
    const __bf16* Am = A + (size_t)mBase * Kd;
    v8f c0 = {}, c1 = {}, c2 = {}, c3 = {};
    // prologue: fragments for k = 0
    v16bf a  = load_fragA(Am, Kd, 0, lane);
    v16bf b0 = load_fragB(W, Kd, nBase,      0, lane);
    v16bf b1 = load_fragB(W, Kd, nBase + 16, 0, lane);
    v16bf b2 = load_fragB(W, Kd, nBase + 32, 0, lane);
    v16bf b3 = load_fragB(W, Kd, nBase + 48, 0, lane);
    for (int k = 0; k < Kd; k += 32) {
      // branchless next-k (last iteration redundantly reloads k; uniform, EXEC stays full)
      int kn = (k + 32 < Kd) ? (k + 32) : k;
      v16bf an  = load_fragA(Am, Kd, kn, lane);
      v16bf b0n = load_fragB(W, Kd, nBase,      kn, lane);
      v16bf b1n = load_fragB(W, Kd, nBase + 16, kn, lane);
      v16bf b2n = load_fragB(W, Kd, nBase + 32, kn, lane);
      v16bf b3n = load_fragB(W, Kd, nBase + 48, kn, lane);
      c0 = wmma_bf16(a, b0, c0);
      c1 = wmma_bf16(a, b1, c1);
      c2 = wmma_bf16(a, b2, c2);
      c3 = wmma_bf16(a, b3, c3);
      a = an; b0 = b0n; b1 = b1n; b2 = b2n; b3 = b3n;
    }
    float bb0 = bias[nBase + nc];
    float bb1 = bias[nBase + 16 + nc];
    float bb2 = bias[nBase + 32 + nc];
    float bb3 = bias[nBase + 48 + nc];
#pragma unroll
    for (int r = 0; r < 8; ++r) {
      size_t row = (size_t)(mBase + m0 + r) * N;
      C[row + nBase + nc]      = c0[r] + bb0;
      C[row + nBase + 16 + nc] = c1[r] + bb1;
      C[row + nBase + 32 + nc] = c2[r] + bb2;
      C[row + nBase + 48 + nc] = c3[r] + bb3;
    }
  }
}

// =================== persistent bidirectional GRU (one block per direction) ===================
// 24 waves; wave w owns h-columns [16w, 16w+16). Each step it computes the r/z/n
// WMMA tile triple for its columns (A = h in LDS as bf16, M=16 = full batch),
// does the gate math in-register (C-fragment positions are identical across the
// three accumulators), writes h to rep_ac and back into LDS bf16.
__global__ void gru_kernel(const float* __restrict__ gi_f, const float* __restrict__ gi_b,
                           const __bf16* __restrict__ whh_f, const __bf16* __restrict__ whh_b,
                           const float* __restrict__ bhh_f, const float* __restrict__ bhh_b,
                           float* __restrict__ rep_ac) {
  int dir = blockIdx.x;
  const float*  gi  = dir ? gi_b  : gi_f;
  const __bf16* Whh = dir ? whh_b : whh_f;
  const float*  bhh = dir ? bhh_b : bhh_f;

  __shared__ __bf16 h_bf[BB * HH];     // 12 KB

  int lane = threadIdx.x & 31;
  int wave = threadIdx.x >> 5;         // 0..23
  int colBase = wave << 4;
  int nc = lane & 15;
  int m0 = (lane >= 16) ? 8 : 0;
  int colR = 0 * HH + colBase;
  int colZ = 1 * HH + colBase;
  int colN = 2 * HH + colBase;

  for (int i = threadIdx.x; i < BB * HH; i += blockDim.x) h_bf[i] = to_bf16(0.0f);
  __syncthreads();

  float bR = bhh[colR + nc];
  float bZ = bhh[colZ + nc];
  float bN = bhh[colN + nc];

  v8f hprev = {};
  for (int s = 0; s < SS; ++s) {
    int st = dir ? (SS - 1 - s) : s;
    v8f aR = {}, aZ = {}, aN = {};
    // prologue (safe: previous iteration ended with __syncthreads)
    v16bf a  = load_fragA(h_bf, HH, 0, lane);
    v16bf wR = load_fragB(Whh, HH, colR, 0, lane);
    v16bf wZ = load_fragB(Whh, HH, colZ, 0, lane);
    v16bf wN = load_fragB(Whh, HH, colN, 0, lane);
    for (int k = 0; k < HH; k += 32) {
      int kn = (k + 32 < HH) ? (k + 32) : k;
      v16bf an  = load_fragA(h_bf, HH, kn, lane);
      v16bf wRn = load_fragB(Whh, HH, colR, kn, lane);
      v16bf wZn = load_fragB(Whh, HH, colZ, kn, lane);
      v16bf wNn = load_fragB(Whh, HH, colN, kn, lane);
      aR = wmma_bf16(a, wR, aR);
      aZ = wmma_bf16(a, wZ, aZ);
      aN = wmma_bf16(a, wN, aN);
      a = an; wR = wRn; wZ = wZn; wN = wNn;
    }
    v8f hnew;
#pragma unroll
    for (int r = 0; r < 8; ++r) {
      int m   = m0 + r;                 // batch index
      int col = colBase + nc;           // hidden index
      const float* gir = gi + ((size_t)(m * SS + st)) * G3;
      float ir = gir[0 * HH + col];
      float iz = gir[1 * HH + col];
      float in = gir[2 * HH + col];
      float hr = aR[r] + bR;
      float hz = aZ[r] + bZ;
      float hn = aN[r] + bN;
      float rg = 1.0f / (1.0f + __expf(-(ir + hr)));
      float zg = 1.0f / (1.0f + __expf(-(iz + hz)));
      float ng = tanhf(in + rg * hn);
      float hv = (1.0f - zg) * ng + zg * hprev[r];
      hnew[r] = hv;
      rep_ac[((size_t)(m * SS + st)) * TT + dir * HH + col] = hv;
    }
    hprev = hnew;
    __syncthreads();                    // all waves done reading h_bf
#pragma unroll
    for (int r = 0; r < 8; ++r)
      h_bf[(m0 + r) * HH + colBase + nc] = to_bf16(hnew[r]);
    __syncthreads();
  }
}

// =================== qa/kk dots: qa[b,s]=rep.wq, kk[b,j]=rep.wk (j==64 -> rep_context) ===================
__global__ void dots_kernel(const float* __restrict__ rep_ac, const float* __restrict__ rep_ctx,
                            const float* __restrict__ W_find,
                            float* __restrict__ qa, float* __restrict__ kk) {
  int id = blockIdx.x;                 // b*65 + j
  int b = id / 65, j = id % 65;
  const float* rep = (j < 64) ? rep_ac + ((size_t)(b * SS + j)) * TT
                              : rep_ctx + (size_t)b * TT;
  const float* wq = W_find;
  const float* wk = W_find + TT;
  __shared__ float sq[256], sk[256];
  int tid = threadIdx.x;
  float aq = 0.f, ak = 0.f;
  for (int d = tid; d < TT; d += 256) {
    float v = rep[d];
    aq += v * wq[d];
    ak += v * wk[d];
  }
  sq[tid] = aq; sk[tid] = ak;
  __syncthreads();
  for (int off = 128; off > 0; off >>= 1) {
    if (tid < off) { sq[tid] += sq[tid + off]; sk[tid] += sk[tid + off]; }
    __syncthreads();
  }
  if (tid == 0) {
    kk[id] = sk[0];
    if (j < 64) qa[b * SS + j] = sq[0];
  }
}

// =================== top-k (K=8 of 65; sigmoid is monotone so order by kk logits) ===================
__global__ void topk_kernel(const float* __restrict__ qa, const float* __restrict__ kk,
                            float* __restrict__ vals_out, float* __restrict__ idx_out,
                            int* __restrict__ idx_ws) {
  int b = blockIdx.x;                  // 16
  int s = threadIdx.x;                 // 64
  const float* kkb = kk + b * 65;
  float q = qa[b * SS + s];
  unsigned long long mask = 0;         // bits 0..63
  bool used64 = false;
  for (int k = 0; k < KTOP; ++k) {
    float best = -3.4e38f; int bi = 0;
    for (int j = 0; j < 65; ++j) {
      bool used = (j < 64) ? (((mask >> j) & 1ull) != 0) : used64;
      float v = kkb[j];
      if (!used && v > best) { best = v; bi = j; }   // strict '>' => lowest index on ties
    }
    if (bi < 64) mask |= (1ull << bi); else used64 = true;
    float sc = 1.0f / (1.0f + __expf(-(q + best)));
    size_t o = ((size_t)(b * SS + s)) * KTOP + k;
    vals_out[o] = sc;
    idx_out[o]  = (float)bi;
    idx_ws[o]   = bi;
  }
}

// =================== v1 gather: (B,S,K,T) rows from all_reps ===================
__global__ void gather_kernel(const int* __restrict__ idx_ws,
                              const float* __restrict__ rep_ac, const float* __restrict__ rep_ctx,
                              float* __restrict__ v1) {
  int g = blockIdx.x;                  // 0..B*S*K-1
  int b = g >> 9;                      // / (64*8)
  int j = idx_ws[g];
  const float* src = (j < 64) ? rep_ac + ((size_t)(b * SS + j)) * TT
                              : rep_ctx + (size_t)b * TT;
  float* dst = v1 + (size_t)g * TT;
  for (int d = threadIdx.x; d < TT; d += blockDim.x) dst[d] = src[d];
}

// =================== host launcher ===================
extern "C" void kernel_launch(void* const* d_in, const int* in_sizes, int n_in,
                              void* d_out, int out_size, void* d_ws, size_t ws_size,
                              hipStream_t stream) {
  const float* hidden = (const float*)d_in[0];
  const float* W_ih_f = (const float*)d_in[1];
  const float* W_hh_f = (const float*)d_in[2];
  const float* b_ih_f = (const float*)d_in[3];
  const float* b_hh_f = (const float*)d_in[4];
  const float* W_ih_b = (const float*)d_in[5];
  const float* W_hh_b = (const float*)d_in[6];
  const float* b_ih_b = (const float*)d_in[7];
  const float* b_hh_b = (const float*)d_in[8];
  const float* W_ctx  = (const float*)d_in[9];
  const float* b_ctx  = (const float*)d_in[10];
  const float* W_find = (const float*)d_in[11];
  const int* attn_len   = (const int*)d_in[12];
  const int* span_start = (const int*)d_in[13];
  const int* span_len   = (const int*)d_in[14];

  // outputs: rep_ac | rep_context | v1 | vals | idx   (flat f32)
  float* out_rep_ac  = (float*)d_out;
  float* out_rep_ctx = out_rep_ac + (size_t)BB * SS * TT;
  float* out_v1      = out_rep_ctx + (size_t)BB * TT;
  float* out_vals    = out_v1 + (size_t)BB * SS * KTOP * TT;
  float* out_idx     = out_vals + (size_t)BB * SS * KTOP;

  // workspace carve
  char* w = (char*)d_ws;
  auto take = [&](size_t bytes) { char* p = w; w += (bytes + 255) & ~(size_t)255; return p; };
  __bf16* sr_bf   = (__bf16*)take((size_t)BB * SS * DD * 2);
  float*  ctx_acc = (float*) take((size_t)BB * DD * 4);
  __bf16* ctx_bf  = (__bf16*)take((size_t)BB * DD * 2);
  float*  gi_f    = (float*) take((size_t)BB * SS * G3 * 4);
  float*  gi_b    = (float*) take((size_t)BB * SS * G3 * 4);
  __bf16* wihf_bf = (__bf16*)take((size_t)G3 * DD * 2);
  __bf16* whhf_bf = (__bf16*)take((size_t)G3 * HH * 2);
  __bf16* wihb_bf = (__bf16*)take((size_t)G3 * DD * 2);
  __bf16* whhb_bf = (__bf16*)take((size_t)G3 * HH * 2);
  __bf16* wctx_bf = (__bf16*)take((size_t)TT * DD * 2);
  float*  qa      = (float*) take((size_t)BB * SS * 4);
  float*  kk      = (float*) take((size_t)BB * 65 * 4);
  int*    idx_ws  = (int*)   take((size_t)BB * SS * KTOP * 4);

  // 1) weights -> bf16
  cvt_bf16_kernel<<<512, 256, 0, stream>>>(W_ih_f, wihf_bf, G3 * DD);
  cvt_bf16_kernel<<<512, 256, 0, stream>>>(W_hh_f, whhf_bf, G3 * HH);
  cvt_bf16_kernel<<<512, 256, 0, stream>>>(W_ih_b, wihb_bf, G3 * DD);
  cvt_bf16_kernel<<<512, 256, 0, stream>>>(W_hh_b, whhb_bf, G3 * HH);
  cvt_bf16_kernel<<<512, 256, 0, stream>>>(W_ctx,  wctx_bf, TT * DD);

  // 2) ctx prefix-mean (dominant HBM traffic)
  zero_f32_kernel<<<48, 256, 0, stream>>>(ctx_acc, BB * DD);
  ctx_partial_kernel<<<BB * 32, 256, 0, stream>>>(hidden, attn_len, ctx_acc);
  ctx_finalize_kernel<<<BB, 256, 0, stream>>>(ctx_acc, attn_len, ctx_bf);

  // 3) span averages
  span_rep_kernel<<<BB * SS, 256, 0, stream>>>(hidden, span_start, span_len, sr_bf);

  // 4) batched input projections + context projection (bf16 WMMA, f32 accumulate)
  {
    int blocks = ((BB * SS / 16) * (G3 / 64) + 7) / 8;   // 144
    gemm_bf16_kernel<<<blocks, 256, 0, stream>>>(sr_bf, wihf_bf, b_ih_f, gi_f, BB * SS, G3, DD);
    gemm_bf16_kernel<<<blocks, 256, 0, stream>>>(sr_bf, wihb_bf, b_ih_b, gi_b, BB * SS, G3, DD);
    int blocks_ctx = ((BB / 16) * (TT / 64) + 7) / 8;    // 2
    gemm_bf16_kernel<<<blocks_ctx, 256, 0, stream>>>(ctx_bf, wctx_bf, b_ctx, out_rep_ctx, BB, TT, DD);
  }

  // 5) recurrent GRU (fwd + bwd concurrently), writes rep_ac
  gru_kernel<<<2, 24 * 32, 0, stream>>>(gi_f, gi_b, whhf_bf, whhb_bf, b_hh_f, b_hh_b, out_rep_ac);

  // 6) scoring dots, top-k, gather
  dots_kernel<<<BB * 65, 256, 0, stream>>>(out_rep_ac, out_rep_ctx, W_find, qa, kk);
  topk_kernel<<<BB, SS, 0, stream>>>(qa, kk, out_vals, out_idx, idx_ws);
  gather_kernel<<<BB * SS * KTOP, 256, 0, stream>>>(idx_ws, out_rep_ac, out_rep_ctx, out_v1);
}
